// rnn_decoder_36395552867054
// MI455X (gfx1250) — matrix operand order repaired and checked by reading, not compiled, not run on hardware
//
#include <hip/hip_runtime.h>
#include <hip/hip_bf16.h>
#include <cstdint>

// Problem dims (fixed by the reference)
#define B_  256
#define T_  256
#define H_  1024
#define E_  512
#define V_  32000
#define BN_ 512
#define NEG_ (-10000000.0f)

typedef __bf16 bf16_t;
typedef __attribute__((ext_vector_type(4)))  __bf16 v4bf;
typedef __attribute__((ext_vector_type(8)))  __bf16 v8bf;
typedef __attribute__((ext_vector_type(16))) __bf16 v16bf;
typedef __attribute__((ext_vector_type(8)))  float  v8f;

__device__ __forceinline__ float fast_sigmoid(float x) {
  return 1.f / (1.f + __expf(-x));
}
__device__ __forceinline__ float fast_tanh(float x) {
  float cx = fminf(fmaxf(x, -15.f), 15.f);
  float e  = __expf(2.f * cx);
  return (e - 1.f) / (e + 1.f);
}

__device__ __forceinline__ v8bf cvt8(const float4 a, const float4 b) {
  v8bf r;
  r[0] = (__bf16)a.x; r[1] = (__bf16)a.y; r[2] = (__bf16)a.z; r[3] = (__bf16)a.w;
  r[4] = (__bf16)b.x; r[5] = (__bf16)b.y; r[6] = (__bf16)b.z; r[7] = (__bf16)b.w;
  return r;
}

// gfx1250 async global->LDS copy, 16B per lane, tracked by ASYNCcnt.
__device__ __forceinline__ void async_copy_b128(unsigned lds_addr,
                                                const void* gaddr) {
  asm volatile("global_load_async_to_lds_b128 %0, %1, off"
               :: "v"(lds_addr),
                  "v"((unsigned long long)(uintptr_t)gaddr)
               : "memory");
}
__device__ __forceinline__ void wait_asynccnt0() {
  asm volatile("s_wait_asynccnt 0x0" ::: "memory");
}
__device__ __forceinline__ unsigned lds_offset(const void* p) {
  return (unsigned)(uintptr_t)p;  // low 32 bits of generic addr = LDS offset
}

// ---------------------------------------------------------------------------
// Weight preparation: f32 [K][N] -> bf16 [N][K]  (32x32 LDS tile transpose)
// ---------------------------------------------------------------------------
__global__ __launch_bounds__(256) void transpose_to_bf16(
    const float* __restrict__ src, int K, int N, bf16_t* __restrict__ dst) {
  __shared__ float tile[32][33];
  const int k0 = blockIdx.y * 32;
  const int n0 = blockIdx.x * 32;
  {
    const int r = threadIdx.x >> 3;
    const int c = (threadIdx.x & 7) << 2;
    float4 f = *(const float4*)&src[(size_t)(k0 + r) * N + n0 + c];
    tile[r][c] = f.x; tile[r][c + 1] = f.y;
    tile[r][c + 2] = f.z; tile[r][c + 3] = f.w;
  }
  __syncthreads();
  {
    const int n = threadIdx.x >> 3;
    const int k = (threadIdx.x & 7) << 2;
    v4bf o;
    o[0] = (__bf16)tile[k][n];     o[1] = (__bf16)tile[k + 1][n];
    o[2] = (__bf16)tile[k + 2][n]; o[3] = (__bf16)tile[k + 3][n];
    *(v4bf*)&dst[(size_t)(n0 + n) * K + k0 + k] = o;
  }
}

// f32 -> bf16 straight convert (for weights already stored [N][K])
__global__ __launch_bounds__(256) void convert_bf16(
    const float* __restrict__ src, bf16_t* __restrict__ dst, int n) {
  const int i = (blockIdx.x * 256 + threadIdx.x) << 2;
  if (i >= n) return;
  float4 f = *(const float4*)&src[i];
  v4bf o;
  o[0] = (__bf16)f.x; o[1] = (__bf16)f.y; o[2] = (__bf16)f.z; o[3] = (__bf16)f.w;
  *(v4bf*)&dst[i] = o;
}

// ---------------------------------------------------------------------------
// WMMA GEMM:  C[M,N] = act( A_f32[M,K] * B_bf16[N][K]^T + bias (+C) )
//   Tile 64(M) x 128(N), K-stage 64 (2 WMMA K-steps). 8 waves in 2x4 grid,
//   each wave 32x32 via 2x2 v_wmma_f32_16x16x32_bf16 fragments.
//   B tile streamed with GLOBAL_LOAD_ASYNC_TO_LDS_B128 (ASYNCcnt).
// ---------------------------------------------------------------------------
template <int ACCUM, int BIAS, int ACT>
__global__ __launch_bounds__(256) void gemm_wmma(
    const float* __restrict__ A, int lda,
    const bf16_t* __restrict__ Bb, int ldb,  // bf16 [N][K]
    float* __restrict__ C, int ldc, int K,
    const float* __restrict__ bias) {
  constexpr int KB = 64;
  constexpr int AROW = 80;  // 64 + 16 pad bf16 elems (160B rows, 16B aligned)
  constexpr int BROW = 80;
  __shared__ __align__(16) bf16_t As[64 * AROW];
  __shared__ __align__(16) bf16_t Bs[128 * BROW];

  const int tid  = threadIdx.x;
  const int lane = tid & 31;
  const int wave = tid >> 5;
  const int wm   = wave >> 2;   // 0..1
  const int wn   = wave & 3;    // 0..3
  const int m0   = blockIdx.y * 64;
  const int n0   = blockIdx.x * 128;
  const int lh   = lane >> 4;
  const int l16  = lane & 15;

  // staging coordinates
  const int ar = tid >> 2;            // 0..63
  const int ac = (tid & 3) << 4;      // 0,16,32,48 (floats)
  const int bn = tid >> 1;            // 0..127
  const int bc = (tid & 1) << 5;      // 0,32 (bf16)
  const float*  apg = A + (size_t)(m0 + ar) * lda + ac;
  const bf16_t* bpg = Bb + (size_t)(n0 + bn) * ldb + bc;
  const unsigned bs_addr = lds_offset(&Bs[bn * BROW + bc]);

  v8f acc[2][2] = {};

  for (int k0 = 0; k0 < K; k0 += KB) {
    __syncthreads();
    {  // B tile 128x64 bf16: async global->LDS, 4x16B per thread
      const bf16_t* bp = bpg + k0;
      async_copy_b128(bs_addr,      bp);
      async_copy_b128(bs_addr + 16, bp + 8);
      async_copy_b128(bs_addr + 32, bp + 16);
      async_copy_b128(bs_addr + 48, bp + 24);
    }
    {  // A tile 64x64 f32 -> bf16 (16 floats/thread, coalesced)
      const float* ap = apg + k0;
      float4 f0 = *(const float4*)ap;
      float4 f1 = *(const float4*)(ap + 4);
      float4 f2 = *(const float4*)(ap + 8);
      float4 f3 = *(const float4*)(ap + 12);
      *(v8bf*)&As[ar * AROW + ac]     = cvt8(f0, f1);
      *(v8bf*)&As[ar * AROW + ac + 8] = cvt8(f2, f3);
    }
    if (k0 + KB < K) {  // prefetch next tiles (global_prefetch_b8)
      __builtin_prefetch(apg + k0 + KB, 0, 3);
      __builtin_prefetch(bpg + k0 + KB, 0, 3);
    }
    wait_asynccnt0();   // async B-tile writes visible in LDS
    __syncthreads();

#pragma unroll
    for (int ks = 0; ks < KB; ks += 32) {
      v16bf af[2], bfr[2];
#pragma unroll
      for (int mi = 0; mi < 2; mi++) {
        const bf16_t* p = &As[(wm * 32 + mi * 16 + l16) * AROW + ks + lh * 8];
        v8bf lo = *(const v8bf*)p;
        v8bf hi = *(const v8bf*)(p + 16);
        af[mi] = __builtin_shufflevector(lo, hi, 0, 1, 2, 3, 4, 5, 6, 7,
                                         8, 9, 10, 11, 12, 13, 14, 15);
      }
#pragma unroll
      for (int ni = 0; ni < 2; ni++) {
        const bf16_t* p = &Bs[(wn * 32 + ni * 16 + l16) * BROW + ks + lh * 8];
        v8bf lo = *(const v8bf*)p;
        v8bf hi = *(const v8bf*)(p + 16);
        bfr[ni] = __builtin_shufflevector(lo, hi, 0, 1, 2, 3, 4, 5, 6, 7,
                                          8, 9, 10, 11, 12, 13, 14, 15);
      }
#pragma unroll
      for (int mi = 0; mi < 2; mi++)
#pragma unroll
        for (int ni = 0; ni < 2; ni++)
          acc[mi][ni] = __builtin_amdgcn_wmma_f32_16x16x32_bf16(
              false, af[mi], false, bfr[ni], (short)0, acc[mi][ni], false,
              false);
    }
  }

  // Epilogue: lane holds col n=l16, rows mb+r (r=0..7)
#pragma unroll
  for (int mi = 0; mi < 2; mi++) {
#pragma unroll
    for (int ni = 0; ni < 2; ni++) {
      const int n  = n0 + wn * 32 + ni * 16 + l16;
      const int mb = m0 + wm * 32 + mi * 16 + lh * 8;
      float bv = 0.f;
      if constexpr (BIAS) bv = bias[n];
      float* cp = C + (size_t)mb * ldc + n;
#pragma unroll
      for (int r = 0; r < 8; r++) {
        float v = acc[mi][ni][r] + bv;
        if constexpr (ACCUM) v += cp[(size_t)r * ldc];
        if constexpr (ACT) v = fast_tanh(v);
        cp[(size_t)r * ldc] = v;
      }
    }
  }
}

// ---------------------------------------------------------------------------
// Reductions
// ---------------------------------------------------------------------------
template <int NT>
__device__ __forceinline__ float block_sum(float v) {
  __shared__ float s[NT / 32];
  __syncthreads();
#pragma unroll
  for (int o = 16; o > 0; o >>= 1) v += __shfl_xor(v, o, 32);
  if ((threadIdx.x & 31) == 0) s[threadIdx.x >> 5] = v;
  __syncthreads();
  if (threadIdx.x == 0) {
    float t = 0.f;
#pragma unroll
    for (int i = 0; i < NT / 32; i++) t += s[i];
    s[0] = t;
  }
  __syncthreads();
  return s[0];
}

template <int NT>
__device__ __forceinline__ float block_max(float v) {
  __shared__ float s[NT / 32];
  __syncthreads();
#pragma unroll
  for (int o = 16; o > 0; o >>= 1) v = fmaxf(v, __shfl_xor(v, o, 32));
  if ((threadIdx.x & 31) == 0) s[threadIdx.x >> 5] = v;
  __syncthreads();
  if (threadIdx.x == 0) {
    float t = -3.4e38f;
#pragma unroll
    for (int i = 0; i < NT / 32; i++) t = fmaxf(t, s[i]);
    s[0] = t;
  }
  __syncthreads();
  return s[0];
}

// ---------------------------------------------------------------------------
// Elementwise / reduction kernels
// ---------------------------------------------------------------------------
__global__ __launch_bounds__(256) void softmax_rows(
    const float* __restrict__ x, float* __restrict__ y, int ncols) {
  const int b = blockIdx.x;
  const float* xr = x + (size_t)b * ncols;
  float* yr = y + (size_t)b * ncols;
  float mx = -3.4e38f;
  for (int j = threadIdx.x; j < ncols; j += blockDim.x) mx = fmaxf(mx, xr[j]);
  mx = block_max<256>(mx);
  float sum = 0.f;
  for (int j = threadIdx.x; j < ncols; j += blockDim.x) {
    float e = __expf(xr[j] - mx);
    yr[j] = e;
    sum += e;
  }
  sum = block_sum<256>(sum);
  const float inv = 1.0f / sum;
  for (int j = threadIdx.x; j < ncols; j += blockDim.x) yr[j] *= inv;
}

__global__ __launch_bounds__(128) void gather_rows(
    const int* __restrict__ ids, const float* __restrict__ emb,
    float* __restrict__ out) {
  const int b = blockIdx.x;
  const size_t src = (size_t)ids[b] * E_;
  for (int j = threadIdx.x * 4; j < E_; j += blockDim.x * 4)
    *(float4*)&out[(size_t)b * E_ + j] = *(const float4*)&emb[src + j];
}

__global__ __launch_bounds__(256) void gate_embs(
    float* __restrict__ embs, const float* __restrict__ avg,
    const float* __restrict__ g, int n) {
  const int i = blockIdx.x * blockDim.x + threadIdx.x;
  if (i >= n) return;
  const float hg = fast_sigmoid(g[i]);
  embs[i] = hg * embs[i] + (1.f - hg) * avg[i];
}

__global__ __launch_bounds__(128) void att_score(
    const float* __restrict__ proj, const float* __restrict__ qwd,
    const float* __restrict__ v, float* __restrict__ scores, int rowbase) {
  const int row  = blockIdx.x;
  const int rowg = rowbase + row;
  const int b    = rowg >> 8;  // T_ == 256
  const float* pr = proj + (size_t)row * H_;
  const float* qr = qwd + (size_t)b * H_;
  float acc = 0.f;
  for (int n = threadIdx.x; n < H_; n += 128)
    acc += fast_tanh(pr[n] + qr[n]) * v[n];
  acc = block_sum<128>(acc);
  if (threadIdx.x == 0) scores[rowg] = acc;
}

__global__ __launch_bounds__(256) void ctx_weight(
    const float* __restrict__ attn, const float* __restrict__ ctx,
    float* __restrict__ ct) {
  const int b = blockIdx.x;
  const int d = blockIdx.y * 256 + threadIdx.x;
  __shared__ float a[T_];
  a[threadIdx.x] = attn[(size_t)b * T_ + threadIdx.x];
  __syncthreads();
  float acc = 0.f;
  const float* cp = ctx + (size_t)b * T_ * (2 * H_) + d;
  for (int t = 0; t < T_; t++) acc += a[t] * cp[(size_t)t * (2 * H_)];
  ct[(size_t)b * (2 * H_) + d] = acc;
}

__global__ __launch_bounds__(256) void gru_combine(
    const float* __restrict__ gi, const float* __restrict__ gh,
    const float* __restrict__ h, float* __restrict__ out) {
  const int i = blockIdx.x * blockDim.x + threadIdx.x;
  if (i >= B_ * H_) return;
  const int b = i / H_, j = i % H_;
  const size_t base = (size_t)b * 3 * H_ + j;
  const float r  = fast_sigmoid(gi[base] + gh[base]);
  const float z  = fast_sigmoid(gi[base + H_] + gh[base + H_]);
  const float nn = fast_tanh(gi[base + 2 * H_] + r * gh[base + 2 * H_]);
  out[i] = (1.f - z) * nn + z * h[i];
}

__global__ __launch_bounds__(256) void copy_cols(
    const float* __restrict__ src, int lds_, float* __restrict__ dst, int ldd,
    int off, int ncol, int rows) {
  const int n = rows * ncol;
  for (int i = blockIdx.x * blockDim.x + threadIdx.x; i < n;
       i += gridDim.x * blockDim.x) {
    const int r = i / ncol, c = i % ncol;
    dst[(size_t)r * ldd + off + c] = src[(size_t)r * lds_ + c];
  }
}

__global__ __launch_bounds__(256) void apply_mask(
    float* __restrict__ pred, const int* __restrict__ mask) {
  const int b = threadIdx.x;  // one row per thread
  float* row = pred + (size_t)b * V_;
  const float pad = row[0];
  const float eos = row[3];
#pragma unroll
  for (int j = 0; j < 8; j++) row[mask[b * 8 + j]] = NEG_;
  row[0] = pad;
  row[3] = eos;
}

// ---------------------------------------------------------------------------
// Launch
// ---------------------------------------------------------------------------
extern "C" void kernel_launch(void* const* d_in, const int* in_sizes, int n_in,
                              void* d_out, int out_size, void* d_ws,
                              size_t ws_size, hipStream_t stream) {
  (void)in_sizes; (void)n_in; (void)out_size; (void)ws_size;
  const int*   input_ids  = (const int*)d_in[0];
  const float* prediction = (const float*)d_in[1];
  const float* context    = (const float*)d_in[2];
  const float* h0         = (const float*)d_in[3];
  const int*   maskp      = (const int*)d_in[4];
  const float* embedding  = (const float*)d_in[5];
  const float* ge1_w = (const float*)d_in[6];
  const float* ge1_b = (const float*)d_in[7];
  const float* ge2_w = (const float*)d_in[8];
  const float* ge2_b = (const float*)d_in[9];
  const float* att_we = (const float*)d_in[10];
  const float* att_wd = (const float*)d_in[11];
  const float* att_v  = (const float*)d_in[12];
  const float* wih0 = (const float*)d_in[13];
  const float* whh0 = (const float*)d_in[14];
  const float* bih0 = (const float*)d_in[15];
  const float* bhh0 = (const float*)d_in[16];
  const float* wih1 = (const float*)d_in[17];
  const float* whh1 = (const float*)d_in[18];
  const float* bih1 = (const float*)d_in[19];
  const float* bhh1 = (const float*)d_in[20];
  const float* bn0_w = (const float*)d_in[21];
  const float* bn0_b = (const float*)d_in[22];
  const float* bn1_w = (const float*)d_in[23];
  const float* bn1_b = (const float*)d_in[24];

  // Output layout: pred (B*V) | h1 (2*B*H) | attn (B*T)
  float* out  = (float*)d_out;
  float* pred = out;
  float* h1_0 = out + (size_t)B_ * V_;
  float* h1_1 = h1_0 + (size_t)B_ * H_;
  float* attn = h1_1 + (size_t)B_ * H_;

  // Workspace carve-up (16B aligned slots)
  char* Wb = (char*)d_ws;
  size_t off = 0;
  auto allocf = [&](size_t n) {
    float* p = (float*)(Wb + off);
    off += ((n * 4 + 15) & ~(size_t)15);
    return p;
  };
  auto allocb = [&](size_t n) {
    bf16_t* p = (bf16_t*)(Wb + off);
    off += ((n * 2 + 15) & ~(size_t)15);
    return p;
  };
  const int IN0 = E_ + 2 * H_;  // 2560
  const int BI  = 3 * H_ + E_;  // 3584

  float* probs   = allocf((size_t)B_ * V_);
  float* embs    = allocf((size_t)B_ * E_);
  float* emb_avg = allocf((size_t)B_ * E_);
  float* gsum    = allocf((size_t)B_ * E_);
  float* qwd     = allocf((size_t)B_ * H_);
  float* scores  = allocf((size_t)B_ * T_);
  float* ct      = allocf((size_t)B_ * 2 * H_);
  float* rnn_in  = allocf((size_t)B_ * IN0);
  float* proj    = allocf((size_t)16 * T_ * H_);  // attention chunk scratch
  float* b0in    = allocf((size_t)B_ * BI);
  float* b0v     = allocf((size_t)B_ * BN_);
  // gi/gh alias proj (disjoint lifetimes: proj only used in attention phase)
  float* gi = proj;
  float* gh = proj + (size_t)B_ * 3 * H_;

  // bf16 weights, all laid out [N][K]
  bf16_t* embT   = allocb((size_t)E_ * V_);         // [512][32000]
  bf16_t* ge1T   = allocb((size_t)E_ * E_);
  bf16_t* ge2T   = allocb((size_t)E_ * E_);
  bf16_t* attWdT = allocb((size_t)H_ * H_);
  bf16_t* attWeT = allocb((size_t)H_ * 2 * H_);     // [1024][2048]
  bf16_t* bn0T   = allocb((size_t)BN_ * BI);        // [512][3584]
  bf16_t* bn1T   = allocb((size_t)V_ * BN_);        // [32000][512]
  bf16_t* wih0B  = allocb((size_t)3 * H_ * IN0);
  bf16_t* whh0B  = allocb((size_t)3 * H_ * H_);
  bf16_t* wih1B  = allocb((size_t)3 * H_ * H_);
  bf16_t* whh1B  = allocb((size_t)3 * H_ * H_);

  // ---- Phase 0: weight conversion (every call; deterministic) ----
  transpose_to_bf16<<<dim3(E_ / 32, V_ / 32), 256, 0, stream>>>(embedding, V_, E_, embT);
  transpose_to_bf16<<<dim3(E_ / 32, E_ / 32), 256, 0, stream>>>(ge1_w, E_, E_, ge1T);
  transpose_to_bf16<<<dim3(E_ / 32, E_ / 32), 256, 0, stream>>>(ge2_w, E_, E_, ge2T);
  transpose_to_bf16<<<dim3(H_ / 32, H_ / 32), 256, 0, stream>>>(att_wd, H_, H_, attWdT);
  transpose_to_bf16<<<dim3(H_ / 32, 2 * H_ / 32), 256, 0, stream>>>(att_we, 2 * H_, H_, attWeT);
  transpose_to_bf16<<<dim3(BN_ / 32, BI / 32), 256, 0, stream>>>(bn0_w, BI, BN_, bn0T);
  transpose_to_bf16<<<dim3(V_ / 32, BN_ / 32), 256, 0, stream>>>(bn1_w, BN_, V_, bn1T);
  convert_bf16<<<(3 * H_ * IN0 / 4 + 255) / 256, 256, 0, stream>>>(wih0, wih0B, 3 * H_ * IN0);
  convert_bf16<<<(3 * H_ * H_ / 4 + 255) / 256, 256, 0, stream>>>(whh0, whh0B, 3 * H_ * H_);
  convert_bf16<<<(3 * H_ * H_ / 4 + 255) / 256, 256, 0, stream>>>(wih1, wih1B, 3 * H_ * H_);
  convert_bf16<<<(3 * H_ * H_ / 4 + 255) / 256, 256, 0, stream>>>(whh1, whh1B, 3 * H_ * H_);

  // ---- Phase 1: embedding mix ----
  softmax_rows<<<B_, 256, 0, stream>>>(prediction, probs, V_);
  gather_rows<<<B_, 128, 0, stream>>>(input_ids, embedding, embs);
  gemm_wmma<0, 0, 0><<<dim3(E_ / 128, B_ / 64), 256, 0, stream>>>(
      probs, V_, embT, V_, emb_avg, E_, V_, nullptr);
  gemm_wmma<0, 1, 0><<<dim3(E_ / 128, B_ / 64), 256, 0, stream>>>(
      embs, E_, ge1T, E_, gsum, E_, E_, ge1_b);
  gemm_wmma<1, 1, 0><<<dim3(E_ / 128, B_ / 64), 256, 0, stream>>>(
      emb_avg, E_, ge2T, E_, gsum, E_, E_, ge2_b);
  gate_embs<<<(B_ * E_ + 255) / 256, 256, 0, stream>>>(embs, emb_avg, gsum,
                                                       B_ * E_);

  // ---- Phase 2: attention ----
  gemm_wmma<0, 0, 0><<<dim3(H_ / 128, B_ / 64), 256, 0, stream>>>(
      h0 + (size_t)B_ * H_, H_, attWdT, H_, qwd, H_, H_, nullptr);
  for (int cb = 0; cb < 16; cb++) {
    const float* Ac = context + (size_t)cb * 16 * T_ * (2 * H_);
    gemm_wmma<0, 0, 0><<<dim3(H_ / 128, (16 * T_) / 64), 256, 0, stream>>>(
        Ac, 2 * H_, attWeT, 2 * H_, proj, H_, 2 * H_, nullptr);
    att_score<<<16 * T_, 128, 0, stream>>>(proj, qwd, att_v, scores,
                                           cb * 16 * T_);
  }
  softmax_rows<<<B_, 256, 0, stream>>>(scores, attn, T_);
  ctx_weight<<<dim3(B_, (2 * H_) / 256), 256, 0, stream>>>(attn, context, ct);

  // ---- Phase 3: GRU stack ----
  copy_cols<<<(B_ * E_ + 255) / 256, 256, 0, stream>>>(embs, E_, rnn_in, IN0,
                                                       0, E_, B_);
  copy_cols<<<(B_ * 2 * H_ + 255) / 256, 256, 0, stream>>>(
      ct, 2 * H_, rnn_in, IN0, E_, 2 * H_, B_);
  gemm_wmma<0, 1, 0><<<dim3(3 * H_ / 128, B_ / 64), 256, 0, stream>>>(
      rnn_in, IN0, wih0B, IN0, gi, 3 * H_, IN0, bih0);
  gemm_wmma<0, 1, 0><<<dim3(3 * H_ / 128, B_ / 64), 256, 0, stream>>>(
      h0, H_, whh0B, H_, gh, 3 * H_, H_, bhh0);
  gru_combine<<<(B_ * H_ + 255) / 256, 256, 0, stream>>>(gi, gh, h0, h1_0);
  gemm_wmma<0, 1, 0><<<dim3(3 * H_ / 128, B_ / 64), 256, 0, stream>>>(
      h1_0, H_, wih1B, H_, gi, 3 * H_, H_, bih1);
  gemm_wmma<0, 1, 0><<<dim3(3 * H_ / 128, B_ / 64), 256, 0, stream>>>(
      h0 + (size_t)B_ * H_, H_, whh1B, H_, gh, 3 * H_, H_, bhh1);
  gru_combine<<<(B_ * H_ + 255) / 256, 256, 0, stream>>>(
      gi, gh, h0 + (size_t)B_ * H_, h1_1);

  // ---- Phase 4: bottleneck + vocab projection ----
  copy_cols<<<(B_ * H_ + 255) / 256, 256, 0, stream>>>(h1_1, H_, b0in, BI, 0,
                                                       H_, B_);
  copy_cols<<<(B_ * 2 * H_ + 255) / 256, 256, 0, stream>>>(
      ct, 2 * H_, b0in, BI, H_, 2 * H_, B_);
  copy_cols<<<(B_ * E_ + 255) / 256, 256, 0, stream>>>(embs, E_, b0in, BI,
                                                       3 * H_, E_, B_);
  gemm_wmma<0, 1, 1><<<dim3(BN_ / 128, B_ / 64), 256, 0, stream>>>(
      b0in, BI, bn0T, BI, b0v, BN_, BI, bn0_b);
  gemm_wmma<0, 1, 0><<<dim3(V_ / 128, B_ / 64), 256, 0, stream>>>(
      b0v, BN_, bn1T, BN_, pred, V_, BN_, bn1_b);
  apply_mask<<<1, 256, 0, stream>>>(pred, maskp);
}
